// HIGNN_model_22136261444228
// MI455X (gfx1250) — matrix operand order repaired and checked by reading, not compiled, not run on hardware
//
#include <hip/hip_runtime.h>

typedef _Float16 v16h __attribute__((ext_vector_type(16)));
typedef _Float16 v8h  __attribute__((ext_vector_type(8)));
typedef float    v8f  __attribute__((ext_vector_type(8)));

#define WAVES_PER_BLOCK 8
#define BLOCK_THREADS   256
#define HIDDEN          64

__device__ __forceinline__ v8f wmma_f16(v16h a, v16h b, v8f c) {
  // 8 args: (neg_a, A, neg_b, B, c_mod, C, reuse_a, reuse_b)
  return __builtin_amdgcn_wmma_f32_16x16x32_f16(false, a, false, b, (short)0, c,
                                                false, false);
}

// Swap values between lane pairs (l, l^16) — 8 packed f16 at a time.
__device__ __forceinline__ v8h shfl_xor16_v8h(v8h v) {
  union { v8h h; int u[4]; } a;
  a.h = v;
#pragma unroll
  for (int i = 0; i < 4; ++i) a.u[i] = __shfl_xor(a.u[i], 16, 32);
  return a.h;
}

// Transposed edge-MLP-scatter kernel (no LDS; all layout conversion via
// half-wave shuffles).
//   din == 3 : d = x[er0[e]] - x[er1[e]],  dst = er1[e]          (2-body / self)
//   din == 6 : d = [x[er1]-x[er0], x[er2]-x[er1]], dst = er2[e]  (3-body)
__global__ void __launch_bounds__(BLOCK_THREADS)
edge_mlp_scatter(const float* __restrict__ x,
                 const int*   __restrict__ er0,
                 const int*   __restrict__ er1,
                 const int*   __restrict__ er2,
                 const float* __restrict__ attr,
                 const float* __restrict__ W1,
                 const float* __restrict__ B1v,
                 const float* __restrict__ W2,
                 const float* __restrict__ B2v,
                 float*       __restrict__ out,
                 int E, int din) {
  const int tid  = threadIdx.x;
  const int lane = tid & 31;
  const int n    = lane & 15;
  const bool hiL = lane >= 16;

  // ---- Constant A fragments, built once, resident in VGPRs. ----
  // Layer 1: A = W1^T tiles (16 hidden rows x 32 K, K = input feature).
  // 16-bit A layout: lanes 0-15 -> M=lane, halves 0-7 = K0..7, 8-15 = K16..23;
  // lanes 16-31 -> M=lane-16, K stripes +8.  All K >= din are zero, so only
  // low lanes / halves < din are populated.
  v16h a1f[4];
#pragma unroll
  for (int t = 0; t < 4; ++t) {
#pragma unroll
    for (int h = 0; h < 16; ++h) {
      const float wv =
          (!hiL && h < din && h < 8) ? W1[h * HIDDEN + t * 16 + n] : 0.0f;
      a1f[t][h] = (_Float16)wv;
    }
  }
  // Layer 2: A = W2^T (16 out-cols x 64 K, K = hidden), two K=32 slices.
  v16h a2f[2];
#pragma unroll
  for (int kt = 0; kt < 2; ++kt) {
#pragma unroll
    for (int h = 0; h < 16; ++h) {
      const int k = kt * 32 + (hiL ? 16 : 0) + h;
      const float wv = (n < 9) ? W2[k * 9 + n] : 0.0f;
      a2f[kt][h] = (_Float16)wv;
    }
  }
  // Biases map onto the D M-index (= accumulator slot v + 8*hiL).
  float bias1r[4][8];
#pragma unroll
  for (int t = 0; t < 4; ++t)
#pragma unroll
    for (int v = 0; v < 8; ++v)
      bias1r[t][v] = B1v[t * 16 + v + (hiL ? 8 : 0)];
  float bias2r[8];
#pragma unroll
  for (int v = 0; v < 8; ++v) {
    const int oc = v + (hiL ? 8 : 0);
    bias2r[v] = (oc < 9) ? B2v[oc] : 0.0f;
  }

  const int nwaves = (gridDim.x * BLOCK_THREADS) >> 5;
  const int gwave  = (blockIdx.x * BLOCK_THREADS + tid) >> 5;
  const int ntiles = (E + 31) >> 5;

  for (int tile = gwave; tile < ntiles; tile += nwaves) {
    const int  e     = tile * 32 + lane;
    const bool valid = e < E;

    // ---- per-lane edge load: gathers + feature diff ----
    float d[6];
#pragma unroll
    for (int i = 0; i < 6; ++i) d[i] = 0.0f;
    float at0 = 0.0f, at1 = 0.0f, at2 = 0.0f;
    int dst = 0;
    if (valid) {
      const int i0 = er0[e];
      const int i1 = er1[e];
      if (din == 3) {
        d[0] = x[3 * i0 + 0] - x[3 * i1 + 0];
        d[1] = x[3 * i0 + 1] - x[3 * i1 + 1];
        d[2] = x[3 * i0 + 2] - x[3 * i1 + 2];
        dst = i1;
      } else {
        const int i2 = er2[e];
        d[0] = x[3 * i1 + 0] - x[3 * i0 + 0];
        d[1] = x[3 * i1 + 1] - x[3 * i0 + 1];
        d[2] = x[3 * i1 + 2] - x[3 * i0 + 2];
        d[3] = x[3 * i2 + 0] - x[3 * i1 + 0];
        d[4] = x[3 * i2 + 1] - x[3 * i1 + 1];
        d[5] = x[3 * i2 + 2] - x[3 * i1 + 2];
        dst = i2;
      }
      at0 = attr[3 * e + 0];
      at1 = attr[3 * e + 1];
      at2 = attr[3 * e + 2];
    }
    // Pack this lane's features as 8 f16 (K = 0..7 of its edge column).
    v8h dh;
#pragma unroll
    for (int i = 0; i < 6; ++i) dh[i] = (_Float16)d[i];
    dh[6] = (_Float16)0.0f;
    dh[7] = (_Float16)0.0f;

    // ---- two 16-edge subtiles per wave iteration ----
#pragma unroll
    for (int sub = 0; sub < 2; ++sub) {
      // B1 = d^T: lane n is edge column n; subtile 1's features live in the
      // high half-wave -> one xor-16 shuffle.
      const v8h dsrc = (sub == 0) ? dh : shfl_xor16_v8h(dh);
      v16h b1frag;
#pragma unroll
      for (int i = 0; i < 16; ++i) b1frag[i] = (_Float16)0.0f;
      if (!hiL) {
#pragma unroll
        for (int i = 0; i < 8; ++i) b1frag[i] = dsrc[i];
      }

      // Layer 1: h^T = W1^T * d^T, 4 hidden tiles, bias in C, ReLU, pack f16.
      v8h hks[4];  // lane's 8 hidden values per tile (M = v + 8*hiL)
#pragma unroll
      for (int t = 0; t < 4; ++t) {
        v8f c;
#pragma unroll
        for (int i = 0; i < 8; ++i) c[i] = bias1r[t][i];
        c = wmma_f16(a1f[t], b1frag, c);
#pragma unroll
        for (int v = 0; v < 8; ++v) {
          const float hv = c[v] > 0.0f ? c[v] : 0.0f;
          hks[t][v] = (_Float16)hv;
        }
      }

      // Layer 2 B fragments: lane n = edge column n, K = hidden 0..63.
      // Each lane keeps its own 8 values and swaps 8 with its partner.
      v8f c2;
#pragma unroll
      for (int i = 0; i < 8; ++i) c2[i] = bias2r[i];
#pragma unroll
      for (int kt = 0; kt < 2; ++kt) {
        const v8h send = hiL ? hks[2 * kt] : hks[2 * kt + 1];
        const v8h recv = shfl_xor16_v8h(send);
        const v8h lo8  = hiL ? recv : hks[2 * kt];       // halves 0-7
        const v8h hi8  = hiL ? hks[2 * kt + 1] : recv;   // halves 8-15
        v16h b2frag;
#pragma unroll
        for (int i = 0; i < 8; ++i) {
          b2frag[i]     = lo8[i];
          b2frag[8 + i] = hi8[i];
        }
        c2 = wmma_f16(a2f[kt], b2frag, c2);
      }

      // out^T: lane n holds out-cols 0-7 of edge n (low) / col 8 (high,v=0).
      const float o8 = __shfl_xor(c2[0], 16, 32);

      // Edge metadata for this subtile (subtile 1 lives in the high half).
      float A0 = at0, A1 = at1, A2v = at2;
      int dn = dst;
      if (sub == 1) {
        A0 = __shfl_xor(at0, 16, 32);
        A1 = __shfl_xor(at1, 16, 32);
        A2v = __shfl_xor(at2, 16, 32);
        dn = __shfl_xor(dst, 16, 32);
      }

      // 3x3 matvec + atomic scatter: lanes 0-15 each own one edge.
      const int eidx = tile * 32 + sub * 16 + lane;
      if (!hiL && eidx < E) {
        const float y0 = c2[0] * A0 + c2[1] * A1 + c2[2] * A2v;
        const float y1 = c2[3] * A0 + c2[4] * A1 + c2[5] * A2v;
        const float y2 = c2[6] * A0 + c2[7] * A1 + o8 * A2v;
        atomicAdd(&out[3 * dn + 0], y0);
        atomicAdd(&out[3 * dn + 1], y1);
        atomicAdd(&out[3 * dn + 2], y2);
      }
    }
  }
}

__global__ void zero_kernel(float* __restrict__ p, int n) {
  const int i = blockIdx.x * blockDim.x + threadIdx.x;
  if (i < n) p[i] = 0.0f;
}

static inline int blocks_for(int E) {
  int b = (E + BLOCK_THREADS - 1) / BLOCK_THREADS;  // 256 edges per block sweep
  if (b < 1) b = 1;
  if (b > 4096) b = 4096;
  return b;
}

extern "C" void kernel_launch(void* const* d_in, const int* in_sizes, int n_in,
                              void* d_out, int out_size, void* d_ws, size_t ws_size,
                              hipStream_t stream) {
  const float* x      = (const float*)d_in[0];
  const int*   e2     = (const int*)d_in[1];
  const int*   e3     = (const int*)d_in[2];
  const int*   es     = (const int*)d_in[3];
  // d_in[4] = edge_1body (unused by reference forward)
  const float* attr2  = (const float*)d_in[5];
  const float* attr3  = (const float*)d_in[6];
  const float* attrs  = (const float*)d_in[7];
  // d_in[8] = edge_attr_1body (unused)
  const float* W1_2b  = (const float*)d_in[9];
  const float* b1_2b  = (const float*)d_in[10];
  const float* W2_2b  = (const float*)d_in[11];
  const float* b2_2b  = (const float*)d_in[12];
  const float* W1_3b  = (const float*)d_in[13];
  const float* b1_3b  = (const float*)d_in[14];
  const float* W2_3b  = (const float*)d_in[15];
  const float* b2_3b  = (const float*)d_in[16];
  const float* W1_s   = (const float*)d_in[17];
  const float* b1_s   = (const float*)d_in[18];
  const float* W2_s   = (const float*)d_in[19];
  const float* b2_s   = (const float*)d_in[20];

  const int E2 = in_sizes[1] / 2;
  const int E3 = in_sizes[2] / 3;
  const int ES = in_sizes[3] / 2;

  float* out = (float*)d_out;

  // d_out is poisoned by the harness; zero it before accumulation.
  zero_kernel<<<(out_size + 255) / 256, 256, 0, stream>>>(out, out_size);

  // 2-body: d = x[j] - x[i], scatter to i = row 1
  edge_mlp_scatter<<<blocks_for(E2), BLOCK_THREADS, 0, stream>>>(
      x, e2, e2 + E2, e2 /*unused*/, attr2,
      W1_2b, b1_2b, W2_2b, b2_2b, out, E2, 3);

  // 3-body: rows j,k,i ; d = [xk-xj, xi-xk], scatter to i = row 2
  edge_mlp_scatter<<<blocks_for(E3), BLOCK_THREADS, 0, stream>>>(
      x, e3, e3 + E3, e3 + 2 * E3, attr3,
      W1_3b, b1_3b, W2_3b, b2_3b, out, E3, 6);

  // 2-body self: same structure as 2-body
  edge_mlp_scatter<<<blocks_for(ES), BLOCK_THREADS, 0, stream>>>(
      x, es, es + ES, es /*unused*/, attrs,
      W1_s, b1_s, W2_s, b2_s, out, ES, 3);
}
